// Spherical_21638045237542
// MI455X (gfx1250) — compile-verified, matrix operand order checked by reading, not run on hardware
//
#include <hip/hip_runtime.h>
#include <hip/hip_bf16.h>

// ---------------------------------------------------------------------------
// Spherical RNN forward for MI455X (gfx1250).
//  - Dense 128-wide GEMM layers: fp32 WMMA (v_wmma_f32_16x16x4_f32)
//  - Batched 128x128 expm (95% of FLOPs): split-precision bf16 WMMA
//    (v_wmma_f32_16x16x32_bf16, K=32/instr = 8x the fp32 WMMA MAC rate;
//     3-product hi/lo scheme keeps fp32-grade accuracy for the rotation scan)
// B=4, T=256, N=H=128, L=2, P=8128
// ---------------------------------------------------------------------------

typedef __attribute__((ext_vector_type(2)))  float        v2f;
typedef __attribute__((ext_vector_type(8)))  float        v8f;
typedef __attribute__((ext_vector_type(16))) __bf16       v16bf;
typedef __attribute__((ext_vector_type(8)))  unsigned int v8u;

#define NB      128
#define MTOT    1024          // B*T
#define PPROJ   8128
#define STRF    132           // fp32 LDS row stride (132 % 64 == 4 -> bank spread)
#define STRB    136           // bf16 LDS row stride (even; 272B/row, 68 dwords, %64==4)

__device__ __forceinline__ v8f wmma4(v2f a, v2f b, v8f c) {
    return __builtin_amdgcn_wmma_f32_16x16x4_f32(
        false, a, false, b, (short)0, c, false, false);
}

__device__ __forceinline__ v8f wmma_bf(v16bf a, v16bf b, v8f c) {
    return __builtin_amdgcn_wmma_f32_16x16x32_bf16(
        false, a, false, b, (short)0, c, false, false);
}

// round-to-nearest-even fp32 -> bf16 (bits)
__device__ __forceinline__ unsigned short f2bf(float f) {
    unsigned u = __float_as_uint(f);
    u += 0x7FFFu + ((u >> 16) & 1u);
    return (unsigned short)(u >> 16);
}
__device__ __forceinline__ float bf2f(unsigned short h) {
    return __uint_as_float(((unsigned)h) << 16);
}

union bfrag { v8u u; v16bf b; };

// ---------------------------------------------------------------------------
// Generic GEMM: C[M,Nout] = X[M,128] * W[Nout,128]^T (+bias) (+leaky relu)
// grid = (Nout/64, M/64), block = 256 (8 wave32); fp32 WMMA.
// dynamic LDS: 2 * 64*132 floats = 67584 B
// ---------------------------------------------------------------------------
__global__ void __launch_bounds__(256)
gemm128_kernel(const float* __restrict__ X, const float* __restrict__ W,
               const float* __restrict__ bias, float* __restrict__ C,
               int Nout, int act) {
    extern __shared__ float smem[];
    float* Xs = smem;               // [64][132]
    float* Ws = smem + 64 * STRF;   // [64][132]

    const int m0 = blockIdx.y * 64;
    const int n0 = blockIdx.x * 64;
    const int tid = threadIdx.x;

#pragma unroll
    for (int it = 0; it < 8; ++it) {
        int idx4 = tid + 256 * it;      // 0..2047 float4 slots
        int row  = idx4 >> 5;
        int c4   = (idx4 & 31) << 2;
        *(float4*)(Xs + row * STRF + c4) =
            *(const float4*)(X + (size_t)(m0 + row) * NB + c4);
        *(float4*)(Ws + row * STRF + c4) =
            *(const float4*)(W + (size_t)(n0 + row) * NB + c4);
    }
    __syncthreads();

    const int wave = tid >> 5;
    const int lane = tid & 31;
    const int half = lane >> 4;
    const int lm   = lane & 15;
    const int tr   = wave >> 1;
    const int tc0  = (wave & 1) * 2;

    v8f acc0 = {0.f,0.f,0.f,0.f,0.f,0.f,0.f,0.f};
    v8f acc1 = {0.f,0.f,0.f,0.f,0.f,0.f,0.f,0.f};

    const float* Ar  = Xs + (tr * 16 + lm) * STRF;
    const float* Br0 = Ws + ((tc0    ) * 16 + lm) * STRF;
    const float* Br1 = Ws + ((tc0 + 1) * 16 + lm) * STRF;

#pragma unroll
    for (int k = 0; k < NB; k += 4) {
        const int ko = k + half * 2;
        v2f a;  a.x  = Ar[ko];  a.y  = Ar[ko + 1];
        v2f b0; b0.x = Br0[ko]; b0.y = Br0[ko + 1];
        v2f b1; b1.x = Br1[ko]; b1.y = Br1[ko + 1];
        acc0 = wmma4(a, b0, acc0);
        acc1 = wmma4(a, b1, acc1);
    }

#pragma unroll
    for (int tcc = 0; tcc < 2; ++tcc) {
        v8f acc = tcc ? acc1 : acc0;
        int n = n0 + (tc0 + tcc) * 16 + lm;
        float bv = bias ? bias[n] : 0.f;
#pragma unroll
        for (int r = 0; r < 8; ++r) {
            int m = m0 + tr * 16 + r + half * 8;
            float v = acc[r] + bv;
            if (act == 1) v = (v > 0.f) ? v : 0.01f * v;
            C[(size_t)m * Nout + n] = v;
        }
    }
}

// ---------------------------------------------------------------------------
// In-LDS 128x128x128 split-precision matmul:
//   C(f32) = (Ah+Al) * M, with M given transposed as (BTh+BTl):
//   C = Ah*Bh + Ah*Bl + Al*Bh   (Al*Bl ~ 2^-16, dropped)
// A* row-major [m][k]; BT* row-major [n][k] (i.e. M transposed), bf16, stride STRB.
// 8 waves, each owns one 16-row strip of C.
// ---------------------------------------------------------------------------
__device__ __forceinline__ void mm128_bf16(float* __restrict__ C,
                                           const unsigned short* __restrict__ Ah,
                                           const unsigned short* __restrict__ Al,
                                           const unsigned short* __restrict__ BTh,
                                           const unsigned short* __restrict__ BTl,
                                           int tid) {
    const int wave  = tid >> 5;
    const int lane  = tid & 31;
    const int khalf = lane >> 4;   // selects K octets {0-7,16-23} vs {8-15,24-31}
    const int lm    = lane & 15;
    const int tr16  = wave * 16;

    const unsigned int* ArH = (const unsigned int*)(Ah + (tr16 + lm) * STRB);
    const unsigned int* ArL = (const unsigned int*)(Al + (tr16 + lm) * STRB);

    for (int tc = 0; tc < 8; ++tc) {
        const unsigned int* BrH = (const unsigned int*)(BTh + (tc * 16 + lm) * STRB);
        const unsigned int* BrL = (const unsigned int*)(BTl + (tc * 16 + lm) * STRB);
        v8f acc = {0.f,0.f,0.f,0.f,0.f,0.f,0.f,0.f};
#pragma unroll
        for (int k = 0; k < NB; k += 32) {
            bfrag ah, al, bh, bl;
#pragma unroll
            for (int v = 0; v < 8; ++v) {
                // ISA 16-bit A layout: VGPR v<4 -> K = 2v + 8*khalf (+k),
                //                      v>=4   -> K = 16 + 2(v-4) + 8*khalf (+k)
                int k0 = k + ((v < 4) ? (2 * v + 8 * khalf)
                                      : (16 + 2 * (v - 4) + 8 * khalf));
                int w = k0 >> 1;   // packed bf16 pair -> dword index
                ah.u[v] = ArH[w];
                al.u[v] = ArL[w];
                bh.u[v] = BrH[w];
                bl.u[v] = BrL[w];
            }
            acc = wmma_bf(ah.b, bh.b, acc);
            acc = wmma_bf(ah.b, bl.b, acc);
            acc = wmma_bf(al.b, bh.b, acc);
        }
#pragma unroll
        for (int r = 0; r < 8; ++r)
            C[(tr16 + r + khalf * 8) * STRF + tc * 16 + lm] = acc[r];
    }
}

// ---------------------------------------------------------------------------
// expm of skew-symmetric 128x128: scaling-and-squaring (s=6) + 8-term Horner.
// One block per (b,t).
// LDS: Mm,Tm fp32 (2*67584 B) + Ah,Al,BTh,BTl bf16 (4*34816 B) = 274432 B
// ---------------------------------------------------------------------------
#define OFF_MM   0
#define OFF_TM   (128 * STRF)                 // floats
#define OFF_AH   (2 * 128 * STRF * 4)         // bytes from base
#define OFF_AL   (OFF_AH + 128 * STRB * 2)
#define OFF_BTH  (OFF_AL + 128 * STRB * 2)
#define OFF_BTL  (OFF_BTH + 128 * STRB * 2)
#define EXPM_LDS (OFF_BTL + 128 * STRB * 2)   // 274432 B

__global__ void __launch_bounds__(256)
expm_kernel(const float* __restrict__ q, float* __restrict__ R) {
    extern __shared__ char smraw[];
    float*          Mm  = (float*)smraw;
    float*          Tm  = (float*)smraw + OFF_TM;
    unsigned short* Ah  = (unsigned short*)(smraw + OFF_AH);
    unsigned short* Al  = (unsigned short*)(smraw + OFF_AL);
    unsigned short* BTh = (unsigned short*)(smraw + OFF_BTH);
    unsigned short* BTl = (unsigned short*)(smraw + OFF_BTL);

    const int bt  = blockIdx.x;
    const int tid = threadIdx.x;
    const float scale = 1.0f / 64.0f;  // 2^-6
    const float* qb = q + (size_t)bt * PPROJ;

    // Build scaled skew matrix directly as bf16 hi/lo (A-side, row major);
    // init M = I in fp32.
    for (int idx = tid; idx < NB * NB; idx += 256) {
        int i = idx >> 7, j = idx & 127;
        float v = 0.f;
        if (i < j)      v =  qb[(i * (255 - i)) / 2 + (j - i - 1)] * scale;
        else if (i > j) v = -qb[(j * (255 - j)) / 2 + (i - j - 1)] * scale;
        unsigned short h = f2bf(v);
        Ah[i * STRB + j] = h;
        Al[i * STRB + j] = f2bf(v - bf2f(h));
        Mm[i * STRF + j] = (i == j) ? 1.f : 0.f;
    }
    __syncthreads();

    // Horner: M <- I + (Bskew*M)/(k+1), k = 7..0  => M = sum_{p<=8} Bskew^p/p!
    for (int k = 7; k >= 0; --k) {
        // right operand: M -> transposed bf16 hi/lo
        for (int idx = tid; idx < NB * NB; idx += 256) {
            int i = idx >> 7, j = idx & 127;     // M[i][j]
            float v = Mm[i * STRF + j];
            unsigned short h = f2bf(v);
            BTh[j * STRB + i] = h;
            BTl[j * STRB + i] = f2bf(v - bf2f(h));
        }
        __syncthreads();
        mm128_bf16(Tm, Ah, Al, BTh, BTl, tid);
        __syncthreads();
        float inv = 1.f / (float)(k + 1);
        for (int idx = tid; idx < NB * NB; idx += 256) {
            int i = idx >> 7, j = idx & 127;
            Mm[i * STRF + j] = Tm[i * STRF + j] * inv + ((i == j) ? 1.f : 0.f);
        }
        __syncthreads();
    }

    // 6 squarings; skew arrays no longer needed -> reuse Ah/Al as left operand
    float* cur = Mm;
    float* oth = Tm;
#pragma unroll
    for (int s = 0; s < 6; ++s) {
        for (int idx = tid; idx < NB * NB; idx += 256) {
            int i = idx >> 7, j = idx & 127;
            float v = cur[i * STRF + j];
            unsigned short h = f2bf(v);
            unsigned short l = f2bf(v - bf2f(h));
            Ah[i * STRB + j]  = h;  Al[i * STRB + j]  = l;   // left, row-major
            BTh[j * STRB + i] = h;  BTl[j * STRB + i] = l;   // right, transposed
        }
        __syncthreads();
        mm128_bf16(oth, Ah, Al, BTh, BTl, tid);
        __syncthreads();
        float* t = cur; cur = oth; oth = t;
    }

    float* Rb = R + (size_t)bt * (NB * NB);
    for (int idx = tid; idx < NB * NB; idx += 256) {
        int i = idx >> 7, j = idx & 127;
        Rb[idx] = cur[i * STRF + j];
    }
}

// ---------------------------------------------------------------------------
// Sequential scan: h_t = R_t @ (start ? h_reset : h_{t-1}); one block per b.
// ---------------------------------------------------------------------------
__global__ void __launch_bounds__(128)
scan_kernel(const float* __restrict__ R, const int* __restrict__ starts,
            const float* __restrict__ h0l, float* __restrict__ hs,
            float* __restrict__ hT) {
    __shared__ float h[NB];
    const int b = blockIdx.x;
    const int i = threadIdx.x;
    const float hres = 0.08838834764831845f;  // 1/sqrt(128)

    h[i] = h0l[b * NB + i];
    __syncthreads();

    for (int t = 0; t < 256; ++t) {
        const size_t bt = (size_t)(b * 256 + t);
        const bool st = (starts[b * 256 + t] != 0);
        const float4* R4 = (const float4*)(R + bt * (NB * NB) + (size_t)i * NB);
        float acc = 0.f;
        if (st) {
#pragma unroll 8
            for (int j4 = 0; j4 < 32; ++j4) {
                float4 r = R4[j4];
                acc += (r.x + r.y + r.z + r.w) * hres;
            }
        } else {
#pragma unroll 8
            for (int j4 = 0; j4 < 32; ++j4) {
                float4 r = R4[j4];
                acc += r.x * h[4 * j4] + r.y * h[4 * j4 + 1] +
                       r.z * h[4 * j4 + 2] + r.w * h[4 * j4 + 3];
            }
        }
        __syncthreads();
        h[i] = acc;
        hs[bt * NB + i] = acc;
        __syncthreads();
    }
    hT[b * NB + i] = h[i];
}

// ---------------------------------------------------------------------------
// host-side launcher
// ---------------------------------------------------------------------------
extern "C" void kernel_launch(void* const* d_in, const int* in_sizes, int n_in,
                              void* d_out, int out_size, void* d_ws, size_t ws_size,
                              hipStream_t stream) {
    const float* x        = (const float*)d_in[0];
    const float* h0       = (const float*)d_in[1];
    const float* map_in_w = (const float*)d_in[2];
    const float* map_in_b = (const float*)d_in[3];
    const float* Wh       = (const float*)d_in[4];
    const float* proj_w   = (const float*)d_in[5];
    const float* proj_b   = (const float*)d_in[6];
    const float* Wy_w     = (const float*)d_in[7];
    const float* Wy_b     = (const float*)d_in[8];
    const float* ff_w     = (const float*)d_in[9];
    const float* ff_b     = (const float*)d_in[10];
    const float* map_out_w= (const float*)d_in[11];
    const float* map_out_b= (const float*)d_in[12];
    const int*   starts   = (const int*)d_in[13];   // bool flags (nonzero = reset)

    float* out_h = (float*)d_out;              // [L=2, B=4, N=128]
    float* out_y = out_h + 2 * 4 * NB;         // [B, T, H]

    char* ws = (char*)d_ws;
    float* R  = (float*)ws; ws += (size_t)MTOT * NB * NB * sizeof(float);  // 64 MB
    float* q  = (float*)ws; ws += (size_t)MTOT * PPROJ * sizeof(float);    // 33.3 MB
    float* z  = (float*)ws; ws += (size_t)MTOT * NB * sizeof(float);
    float* z2 = (float*)ws; ws += (size_t)MTOT * NB * sizeof(float);
    float* u  = (float*)ws; ws += (size_t)MTOT * NB * sizeof(float);
    float* hsb= (float*)ws; ws += (size_t)MTOT * NB * sizeof(float);
    float* y  = (float*)ws; ws += (size_t)MTOT * NB * sizeof(float);

    const size_t gemm_lds = (size_t)2 * 64 * STRF * sizeof(float);   // 67584 B
    const size_t expm_lds = (size_t)EXPM_LDS;                        // 274432 B
    (void)hipFuncSetAttribute((const void*)expm_kernel,
                              hipFuncAttributeMaxDynamicSharedMemorySize, (int)expm_lds);
    (void)hipFuncSetAttribute((const void*)gemm128_kernel,
                              hipFuncAttributeMaxDynamicSharedMemorySize, (int)gemm_lds);

    dim3 blk(256);
    dim3 g128(NB / 64, MTOT / 64);        // (2,16)
    dim3 gproj(PPROJ / 64, MTOT / 64);    // (127,16)

    gemm128_kernel<<<g128, blk, gemm_lds, stream>>>(x, map_in_w, map_in_b, z, NB, 0);

    for (int l = 0; l < 2; ++l) {
        const float* Wh_l  = Wh     + (size_t)l * NB * NB;
        const float* pw_l  = proj_w + (size_t)l * PPROJ * NB;
        const float* pb_l  = proj_b + (size_t)l * PPROJ;
        const float* Wyw_l = Wy_w   + (size_t)l * NB * NB;
        const float* Wyb_l = Wy_b   + (size_t)l * NB;
        const float* ffw_l = ff_w   + (size_t)l * NB * NB;
        const float* ffb_l = ff_b   + (size_t)l * NB;
        const float* h0_l  = h0     + (size_t)l * 4 * NB;

        gemm128_kernel<<<g128, blk, gemm_lds, stream>>>(z, Wh_l, nullptr, u, NB, 0);
        gemm128_kernel<<<gproj, blk, gemm_lds, stream>>>(u, pw_l, pb_l, q, PPROJ, 0);
        expm_kernel<<<dim3(MTOT), blk, expm_lds, stream>>>(q, R);
        scan_kernel<<<dim3(4), dim3(128), 0, stream>>>(R, starts, h0_l, hsb,
                                                       out_h + (size_t)l * 4 * NB);
        gemm128_kernel<<<g128, blk, gemm_lds, stream>>>(hsb, Wyw_l, Wyb_l, y, NB, 0);
        gemm128_kernel<<<g128, blk, gemm_lds, stream>>>(y, ffw_l, ffb_l, z2, NB, 1);
        float* t = z; z = z2; z2 = t;
    }

    gemm128_kernel<<<g128, blk, gemm_lds, stream>>>(z, map_out_w, map_out_b, out_y, NB, 0);
}
